// MultiHeadAttention_50044958933002
// MI455X (gfx1250) — matrix operand order repaired and checked by reading, not compiled, not run on hardware
//
#include <hip/hip_runtime.h>
#include <hip/hip_bf16.h>

// Problem constants (from reference)
#define BB    2
#define TT    2048
#define DD    512
#define HH    8
#define KSZ   64
#define HK    512     // H*K
#define NMODEL 512

typedef __attribute__((ext_vector_type(16))) __bf16 v16bf;
typedef __attribute__((ext_vector_type(8)))  float  v8f;
typedef int async_i4 __attribute__((vector_size(16)));   // matches builtin param type

union Frag16 { v16bf v; uint4 q[2]; };
union Pack8  { uint4 q; __bf16 h[8]; };

#if defined(__has_builtin)
#  if __has_builtin(__builtin_amdgcn_global_load_async_to_lds_b128) && \
      __has_builtin(__builtin_amdgcn_s_wait_asynccnt)
#    define USE_ASYNC_LDS 1
#  endif
#endif
#ifndef USE_ASYNC_LDS
#  define USE_ASYNC_LDS 0
#endif

__device__ __forceinline__ unsigned short f2bf_u(float f) {
    unsigned u = __float_as_uint(f);
    unsigned r = (u + 0x7FFFu + ((u >> 16) & 1u)) >> 16;  // RNE
    return (unsigned short)r;
}
__device__ __forceinline__ __bf16 f2bf(float f) {
    union { unsigned short s; __bf16 b; } cv; cv.s = f2bf_u(f); return cv.b;
}
__device__ __forceinline__ unsigned pack2(float a, float b) {
    return (unsigned)f2bf_u(a) | ((unsigned)f2bf_u(b) << 16);
}

// ---------------------------------------------------------------------------
// Kernel 1: Y(bf16)[4096,512] = X(f32)[4096,512] @ W(f32)[512,512] + bias
// Block tile 128(M) x 64(N), 8 waves, each wave 32x32 = 2x2 WMMA tiles.
// ---------------------------------------------------------------------------
__global__ void __launch_bounds__(256)
proj_f32_bf16(const float* __restrict__ X, const float* __restrict__ W,
              const float* __restrict__ bias, __bf16* __restrict__ Y)
{
    __shared__ __align__(16) __bf16 ldsA[128 * 32];   // M x K, row major
    __shared__ __align__(16) __bf16 ldsBT[64 * 32];   // N x K (transposed)

    const int tid  = threadIdx.x;
    const int lane = tid & 31;
    const int wave = tid >> 5;
    const int half = lane >> 4;
    const int l16  = lane & 15;

    const int blockM = blockIdx.x * 128;
    const int blockN = blockIdx.y * 64;
    const int wm = (wave & 3) * 32;
    const int wn = (wave >> 2) * 32;

    v8f acc[2][2];
    #pragma unroll
    for (int i = 0; i < 2; ++i)
        #pragma unroll
        for (int j = 0; j < 2; ++j)
            acc[i][j] = (v8f){0.f,0.f,0.f,0.f,0.f,0.f,0.f,0.f};

    for (int k0 = 0; k0 < DD; k0 += 32) {
        #pragma unroll
        for (int i = 0; i < 4; ++i) {
            int idx = tid + 256 * i;
            int row = idx >> 3;
            int seg = idx & 7;
            float4 f = *(const float4*)(X + (size_t)(blockM + row) * DD + k0 + seg * 4);
            uint2 p; p.x = pack2(f.x, f.y); p.y = pack2(f.z, f.w);
            *(uint2*)(ldsA + row * 32 + seg * 4) = p;
        }
        {
            int k  = tid & 31;
            int ng = tid >> 5;
            const float* wp = W + (size_t)(k0 + k) * NMODEL + blockN + ng * 8;
            float4 f0 = *(const float4*)(wp);
            float4 f1 = *(const float4*)(wp + 4);
            __bf16* dst = ldsBT + (ng * 8) * 32 + k;
            dst[0*32] = f2bf(f0.x); dst[1*32] = f2bf(f0.y);
            dst[2*32] = f2bf(f0.z); dst[3*32] = f2bf(f0.w);
            dst[4*32] = f2bf(f1.x); dst[5*32] = f2bf(f1.y);
            dst[6*32] = f2bf(f1.z); dst[7*32] = f2bf(f1.w);
        }
        __syncthreads();

        Frag16 a[2], b[2];
        #pragma unroll
        for (int i = 0; i < 2; ++i) {
            const __bf16* p = ldsA + (wm + i * 16 + l16) * 32 + half * 8;
            a[i].q[0] = *(const uint4*)(p);
            a[i].q[1] = *(const uint4*)(p + 16);
        }
        #pragma unroll
        for (int j = 0; j < 2; ++j) {
            const __bf16* p = ldsBT + (wn + j * 16 + l16) * 32 + half * 8;
            b[j].q[0] = *(const uint4*)(p);
            b[j].q[1] = *(const uint4*)(p + 16);
        }
        #pragma unroll
        for (int i = 0; i < 2; ++i)
            #pragma unroll
            for (int j = 0; j < 2; ++j)
                acc[i][j] = __builtin_amdgcn_wmma_f32_16x16x32_bf16(
                    false, a[i].v, false, b[j].v, (short)0, acc[i][j], false, false);
        __syncthreads();
    }

    #pragma unroll
    for (int i = 0; i < 2; ++i)
        #pragma unroll
        for (int j = 0; j < 2; ++j) {
            int col = blockN + wn + j * 16 + l16;
            float bv = bias[col];
            #pragma unroll
            for (int r = 0; r < 8; ++r) {
                int row = blockM + wm + i * 16 + r + 8 * half;
                Y[(size_t)row * HK + col] = f2bf(acc[i][j][r] + bv);
            }
        }
}

// ---------------------------------------------------------------------------
// Kernel 2: flash attention per (b, h). Wave = 16 query rows, block = 128
// query rows; key block = 64 (16 WMMAs per stage, 2 barriers per 64 keys).
// K tile staged via async global->LDS when the builtin exists.
// ---------------------------------------------------------------------------
__global__ void __launch_bounds__(256)
flash_attn(const __bf16* __restrict__ qh, const __bf16* __restrict__ kh,
           const __bf16* __restrict__ vh, const unsigned char* __restrict__ mask,
           __bf16* __restrict__ attn)
{
    __shared__ __align__(16) __bf16 ldsK[64 * 64];      // 64 keys x 64 dims
    __shared__ __align__(16) __bf16 ldsVT[64 * 64];     // 64 dims x 64 keys
    __shared__ __align__(16) __bf16 ldsP[8][16 * 64];   // per-wave P transpose scratch

    const int tid  = threadIdx.x;
    const int lane = tid & 31;
    const int wave = tid >> 5;
    const int half = lane >> 4;
    const int l16  = lane & 15;

    const int b = blockIdx.z;
    const int h = blockIdx.y;
    const int qBase = blockIdx.x * 128 + wave * 16;

    const int  hh      = (h < HH / 2) ? h : h - HH / 2;
    const bool fwdHead = (h < HH / 2);
    const float slope  = __powf(24.0f, -0.25f * (float)(hh + 1));

    // Q fragments (A layout), 16 rows x 64 dims -> two K=32 fragments
    Frag16 aq[2];
    {
        const __bf16* qrow = qh + (size_t)(b * TT + qBase + l16) * HK + h * KSZ + half * 8;
        #pragma unroll
        for (int f = 0; f < 2; ++f) {
            aq[f].q[0] = *(const uint4*)(qrow + f * 32);
            aq[f].q[1] = *(const uint4*)(qrow + f * 32 + 16);
        }
    }

    v8f oacc[4];
    #pragma unroll
    for (int j = 0; j < 4; ++j) oacc[j] = (v8f){0.f,0.f,0.f,0.f,0.f,0.f,0.f,0.f};
    float mrow[8], lrow[8];
    #pragma unroll
    for (int r = 0; r < 8; ++r) { mrow[r] = -1e30f; lrow[r] = 0.0f; }

    const size_t maskBase = (size_t)b * TT * TT;

    for (int j0 = 0; j0 < TT; j0 += 64) {
        // Prefetch next K/V block (global_prefetch_b8)
        if (j0 + 64 < TT) {
            size_t nb = (size_t)(b * TT + j0 + 64 + (tid >> 2)) * HK + h * KSZ + (tid & 3) * 16;
            __builtin_prefetch(kh + nb, 0, 1);
            __builtin_prefetch(vh + nb, 0, 1);
        }
        __syncthreads();   // previous iteration's LDS reads done

        // Stage K (row major) and V^T. 512 16B-chunks over 256 threads.
        #pragma unroll
        for (int i = 0; i < 2; ++i) {
            int idx = tid + 256 * i;
            int row = idx >> 3;         // 0..63 key row
            int seg = idx & 7;          // 8 bf16 per segment
            size_t src = (size_t)(b * TT + j0 + row) * HK + h * KSZ + seg * 8;
#if USE_ASYNC_LDS
            __builtin_amdgcn_global_load_async_to_lds_b128(
                (__attribute__((address_space(1))) async_i4*)(kh + src),
                (__attribute__((address_space(3))) async_i4*)(ldsK + row * 64 + seg * 8),
                0, 0);
#else
            *(uint4*)(ldsK + row * 64 + seg * 8) = *(const uint4*)(kh + src);
#endif
            Pack8 pv; pv.q = *(const uint4*)(vh + src);
            #pragma unroll
            for (int e = 0; e < 8; ++e)
                ldsVT[(seg * 8 + e) * 64 + row] = pv.h[e];
        }
#if USE_ASYNC_LDS
        __builtin_amdgcn_s_wait_asynccnt(0);
#endif
        __syncthreads();

        // S tiles: 16 q-rows x 64 keys (four 16-wide tiles), K-dim = 64 dims
        v8f sT[4];
        #pragma unroll
        for (int nt = 0; nt < 4; ++nt) {
            Frag16 bk0, bk1;
            const __bf16* kp = ldsK + (nt * 16 + l16) * 64 + half * 8;
            bk0.q[0] = *(const uint4*)(kp);       bk0.q[1] = *(const uint4*)(kp + 16);
            bk1.q[0] = *(const uint4*)(kp + 32);  bk1.q[1] = *(const uint4*)(kp + 48);
            v8f s = (v8f){0.f,0.f,0.f,0.f,0.f,0.f,0.f,0.f};
            s = __builtin_amdgcn_wmma_f32_16x16x32_bf16(false, aq[0].v, false, bk0.v, (short)0, s, false, false);
            s = __builtin_amdgcn_wmma_f32_16x16x32_bf16(false, aq[1].v, false, bk1.v, (short)0, s, false, false);
            sT[nt] = s;
        }

        // scale + ALiBi + mask (C layout: row = r + 8*half, col = nt*16 + l16)
        #pragma unroll
        for (int nt = 0; nt < 4; ++nt) {
            #pragma unroll
            for (int r = 0; r < 8; ++r) {
                int mRow = qBase + r + 8 * half;
                int n = j0 + nt * 16 + l16;
                float bias;
                if (fwdHead) bias = (n <= mRow) ? slope * (float)(n - mRow) : -1e9f;
                else         bias = (n >= mRow) ? slope * (float)(mRow - n) : -1e9f;
                float lg = sT[nt][r] * 0.125f + bias;
                if (!mask[maskBase + (size_t)mRow * TT + n]) lg = -1e30f;
                sT[nt][r] = lg;
            }
        }

        // Online softmax per row; write P (bf16) to per-wave LDS for transpose
        #pragma unroll
        for (int r = 0; r < 8; ++r) {
            float v = fmaxf(fmaxf(sT[0][r], sT[1][r]), fmaxf(sT[2][r], sT[3][r]));
            #pragma unroll
            for (int off = 8; off >= 1; off >>= 1)
                v = fmaxf(v, __shfl_xor(v, off, 16));
            float mnew  = fmaxf(mrow[r], v);
            float alpha = __expf(mrow[r] - mnew);
            float p[4], ps = 0.0f;
            #pragma unroll
            for (int nt = 0; nt < 4; ++nt) { p[nt] = __expf(sT[nt][r] - mnew); ps += p[nt]; }
            #pragma unroll
            for (int off = 8; off >= 1; off >>= 1)
                ps += __shfl_xor(ps, off, 16);
            lrow[r] = lrow[r] * alpha + ps;
            mrow[r] = mnew;
            #pragma unroll
            for (int j = 0; j < 4; ++j) oacc[j][r] *= alpha;
            int m = r + 8 * half;
            #pragma unroll
            for (int nt = 0; nt < 4; ++nt)
                ldsP[wave][m * 64 + nt * 16 + l16] = f2bf(p[nt]);
        }

        // Two A-fragments of P (16x64) and V B-fragments (N = 64 head dims)
        Frag16 ap[2];
        #pragma unroll
        for (int f = 0; f < 2; ++f) {
            const __bf16* pp = &ldsP[wave][l16 * 64 + f * 32 + half * 8];
            ap[f].q[0] = *(const uint4*)(pp);
            ap[f].q[1] = *(const uint4*)(pp + 16);
        }
        #pragma unroll
        for (int j = 0; j < 4; ++j) {
            Frag16 bv0, bv1;
            const __bf16* vp = ldsVT + (j * 16 + l16) * 64 + half * 8;
            bv0.q[0] = *(const uint4*)(vp);       bv0.q[1] = *(const uint4*)(vp + 16);
            bv1.q[0] = *(const uint4*)(vp + 32);  bv1.q[1] = *(const uint4*)(vp + 48);
            oacc[j] = __builtin_amdgcn_wmma_f32_16x16x32_bf16(
                false, ap[0].v, false, bv0.v, (short)0, oacc[j], false, false);
            oacc[j] = __builtin_amdgcn_wmma_f32_16x16x32_bf16(
                false, ap[1].v, false, bv1.v, (short)0, oacc[j], false, false);
        }
    }

    #pragma unroll
    for (int j = 0; j < 4; ++j)
        #pragma unroll
        for (int r = 0; r < 8; ++r) {
            int mRow = qBase + r + 8 * half;
            float o = oacc[j][r] / fmaxf(lrow[r], 1e-30f);
            attn[(size_t)(b * TT + mRow) * HK + h * KSZ + j * 16 + l16] = f2bf(o);
        }
}

// ---------------------------------------------------------------------------
// Kernel 3: out(f32)[4096,512] = attn(bf16)[4096,512] @ Wo(f32)[512,512] + bo
// ---------------------------------------------------------------------------
__global__ void __launch_bounds__(256)
out_proj(const __bf16* __restrict__ X, const float* __restrict__ W,
         const float* __restrict__ bias, float* __restrict__ out)
{
    __shared__ __align__(16) __bf16 ldsA[128 * 32];
    __shared__ __align__(16) __bf16 ldsBT[64 * 32];

    const int tid  = threadIdx.x;
    const int lane = tid & 31;
    const int wave = tid >> 5;
    const int half = lane >> 4;
    const int l16  = lane & 15;

    const int blockM = blockIdx.x * 128;
    const int blockN = blockIdx.y * 64;
    const int wm = (wave & 3) * 32;
    const int wn = (wave >> 2) * 32;

    v8f acc[2][2];
    #pragma unroll
    for (int i = 0; i < 2; ++i)
        #pragma unroll
        for (int j = 0; j < 2; ++j)
            acc[i][j] = (v8f){0.f,0.f,0.f,0.f,0.f,0.f,0.f,0.f};

    for (int k0 = 0; k0 < HK; k0 += 32) {
        #pragma unroll
        for (int i = 0; i < 2; ++i) {
            int idx = tid + 256 * i;        // 512 uint4 = 128 rows x 4 segs
            int row = idx >> 2;
            int seg = idx & 3;
            *(uint4*)(ldsA + row * 32 + seg * 8) =
                *(const uint4*)(X + (size_t)(blockM + row) * HK + k0 + seg * 8);
        }
        {
            int k  = tid & 31;
            int ng = tid >> 5;
            const float* wp = W + (size_t)(k0 + k) * NMODEL + blockN + ng * 8;
            float4 f0 = *(const float4*)(wp);
            float4 f1 = *(const float4*)(wp + 4);
            __bf16* dst = ldsBT + (ng * 8) * 32 + k;
            dst[0*32] = f2bf(f0.x); dst[1*32] = f2bf(f0.y);
            dst[2*32] = f2bf(f0.z); dst[3*32] = f2bf(f0.w);
            dst[4*32] = f2bf(f1.x); dst[5*32] = f2bf(f1.y);
            dst[6*32] = f2bf(f1.z); dst[7*32] = f2bf(f1.w);
        }
        __syncthreads();

        Frag16 a[2], b[2];
        #pragma unroll
        for (int i = 0; i < 2; ++i) {
            const __bf16* p = ldsA + (wm + i * 16 + l16) * 32 + half * 8;
            a[i].q[0] = *(const uint4*)(p);
            a[i].q[1] = *(const uint4*)(p + 16);
        }
        #pragma unroll
        for (int j = 0; j < 2; ++j) {
            const __bf16* p = ldsBT + (wn + j * 16 + l16) * 32 + half * 8;
            b[j].q[0] = *(const uint4*)(p);
            b[j].q[1] = *(const uint4*)(p + 16);
        }
        #pragma unroll
        for (int i = 0; i < 2; ++i)
            #pragma unroll
            for (int j = 0; j < 2; ++j)
                acc[i][j] = __builtin_amdgcn_wmma_f32_16x16x32_bf16(
                    false, a[i].v, false, b[j].v, (short)0, acc[i][j], false, false);
        __syncthreads();
    }

    #pragma unroll
    for (int i = 0; i < 2; ++i)
        #pragma unroll
        for (int j = 0; j < 2; ++j) {
            int col = blockN + wn + j * 16 + l16;
            float bv = bias[col];
            #pragma unroll
            for (int r = 0; r < 8; ++r) {
                int row = blockM + wm + i * 16 + r + 8 * half;
                out[(size_t)row * NMODEL + col] = acc[i][j][r] + bv;
            }
        }
}

// ---------------------------------------------------------------------------
extern "C" void kernel_launch(void* const* d_in, const int* in_sizes, int n_in,
                              void* d_out, int out_size, void* d_ws, size_t ws_size,
                              hipStream_t stream) {
    const float* query = (const float*)d_in[0];
    const float* key_  = (const float*)d_in[1];
    const float* value = (const float*)d_in[2];
    const unsigned char* mask = (const unsigned char*)d_in[3];  // numpy bool, 1 byte
    const float* Wq = (const float*)d_in[4];
    const float* bq = (const float*)d_in[5];
    const float* Wk = (const float*)d_in[6];
    const float* bk = (const float*)d_in[7];
    const float* Wv = (const float*)d_in[8];
    const float* bv = (const float*)d_in[9];
    const float* Wo = (const float*)d_in[10];
    const float* bo = (const float*)d_in[11];
    float* out = (float*)d_out;

    const size_t planeBytes = (size_t)BB * TT * HK * sizeof(unsigned short); // 4 MB
    __bf16* qhp = (__bf16*)d_ws;
    __bf16* khp = (__bf16*)((char*)d_ws + planeBytes);
    __bf16* vhp = (__bf16*)((char*)d_ws + 2 * planeBytes);
    __bf16* atp = (__bf16*)((char*)d_ws + 3 * planeBytes);

    dim3 gProj(32, 8);  // (B*T)/128, (H*K)/64
    proj_f32_bf16<<<gProj, 256, 0, stream>>>(query, Wq, bq, qhp);
    proj_f32_bf16<<<gProj, 256, 0, stream>>>(key_,  Wk, bk, khp);
    proj_f32_bf16<<<gProj, 256, 0, stream>>>(value, Wv, bv, vhp);

    dim3 gAttn(TT / 128, HH, BB);
    flash_attn<<<gAttn, 256, 0, stream>>>(qhp, khp, vhp, mask, atp);

    dim3 gOut(32, 8);   // (B*T)/128, MODEL/64
    out_proj<<<gOut, 256, 0, stream>>>(atp, Wo, bo, out);
}
